// LorentzMoE_68289980007144
// MI455X (gfx1250) — compile-verified, matrix operand order checked by reading, not run on hardware
//
#include <hip/hip_runtime.h>
#include <math.h>

// ---------------------------------------------------------------------------
// LorentzMoE forward for gfx1250 (MI455X), wave32 + v_wmma_f32_16x16x32_bf16.
// Grouped-GEMM-by-expert formulation. Weights are pre-converted once per
// launch into zero-padded bf16 workspace (stays L2-resident: 192MB L2 >> 57MB)
// so the GEMM inner loops are pure b128 loads -> WMMA with no conversion VALU
// and no guards. Falls back to in-register f32->bf16 conversion if ws_size is
// too small for the staged layout.
// ---------------------------------------------------------------------------

typedef __bf16 bf16_t;
typedef __attribute__((ext_vector_type(16))) __bf16 v16bf;
typedef __attribute__((ext_vector_type(8)))  __bf16 v8bf;
typedef __attribute__((ext_vector_type(8)))  float   v8f;
typedef __attribute__((ext_vector_type(4)))  float   v4f;

#define T_TOK   512      // tokens
#define DIMF    1024     // full Lorentz dim
#define NEXP    16
#define RH      512      // routed hidden width (time + 511)
#define SSP     1023     // space dim of output / shared inter space
#define SH      1024     // shared hidden width (time + 1023)

#define WMMA_BF16(A,B,C) \
  __builtin_amdgcn_wmma_f32_16x16x32_bf16(false,(A),false,(B),(short)0,(C),false,false)

// Fragment loader (A or B, 16-bit, K=32): per ISA 7.12.2, each lane holds two
// contiguous 8-element K-chunks at offsets kb and kb+16, kb=(lane<16?0:8).
static __device__ __forceinline__ v16bf load_frag_bf16(const bf16_t* __restrict__ p) {
  v8bf c0 = *(const v8bf*)(p);
  v8bf c1 = *(const v8bf*)(p + 16);
  return __builtin_shufflevector(c0, c1, 0, 1, 2, 3, 4, 5, 6, 7,
                                          8, 9, 10, 11, 12, 13, 14, 15);
}

// B fragment from f32 weights (fallback path): packed cvt in-register.
static __device__ __forceinline__ v16bf load_b_f32(const float* __restrict__ p) {
  v8f w0 = *(const v8f*)(p);        // k0 .. k0+7
  v8f w1 = *(const v8f*)(p + 16);   // k0+16 .. k0+23
  v8bf b0 = __builtin_convertvector(w0, v8bf);
  v8bf b1 = __builtin_convertvector(w1, v8bf);
  return __builtin_shufflevector(b0, b1, 0, 1, 2, 3, 4, 5, 6, 7,
                                          8, 9, 10, 11, 12, 13, 14, 15);
}

// ---------------------------------------------------------------------------
// K0: x (f32) -> bf16, 8 elements per thread.
__global__ __launch_bounds__(256)
void k_cvt(const float* __restrict__ x, bf16_t* __restrict__ xb) {
  int i = (blockIdx.x * 256 + threadIdx.x) * 8;
  v8f v = *(const v8f*)(x + i);
  *(v8bf*)(xb + i) = __builtin_convertvector(v, v8bf);
}

// Weight staging: f32 (nsp x KD rows per expert) -> bf16 zero-padded to NPAD
// rows. Output row stride KD; rows >= nsp written as zeros (kills all N-tail
// guards in the GEMM loops). 8 elements per thread.
template <int NPAD, int KD>
__global__ __launch_bounds__(256)
void k_cvt_w(const float* __restrict__ W, bf16_t* __restrict__ Wb, int nsp) {
  size_t idx = ((size_t)blockIdx.x * 256 + threadIdx.x) * 8;
  int e   = (int)(idx / ((size_t)NPAD * KD));
  int rem = (int)(idx % ((size_t)NPAD * KD));
  int row = rem / KD;
  int col = rem % KD;
  v8bf out;
  if (row < nsp) {
    v8f v = *(const v8f*)(W + ((size_t)e * nsp + row) * KD + col);
    out = __builtin_convertvector(v, v8bf);
  } else {
    out = (v8bf)(bf16_t)(0.0f);
  }
  *(v8bf*)(Wb + idx) = out;
}

// ---------------------------------------------------------------------------
// K1: gate -- softmax over 16 experts, top-2, counting-sort tokens into
// per-expert buckets. k=0 slot is the smaller expert index (matches the
// reference's sequential expert scan order for the nonlinear LResNet fold).
__global__ __launch_bounds__(512)
void k_gate(const float* __restrict__ x, const float* __restrict__ gw,
            int* __restrict__ cnt, int* __restrict__ btok,
            int* __restrict__ bdst, float* __restrict__ tokw) {
  int t = threadIdx.x;
  if (t < NEXP) cnt[t] = 0;
  __syncthreads();

  float logit[NEXP];
#pragma unroll
  for (int e = 0; e < NEXP; ++e) logit[e] = 0.f;
  const float* xr = x + (size_t)t * DIMF;
  for (int k = 0; k < DIMF; k += 4) {
    v4f xv = *(const v4f*)(xr + k);
#pragma unroll
    for (int e = 0; e < NEXP; ++e) {
      v4f gv = *(const v4f*)(gw + e * DIMF + k);
      logit[e] += xv[0] * gv[0] + xv[1] * gv[1] + xv[2] * gv[2] + xv[3] * gv[3];
    }
  }
  float mx = logit[0];
#pragma unroll
  for (int e = 1; e < NEXP; ++e) mx = fmaxf(mx, logit[e]);
  float p[NEXP];
  float sum = 0.f;
#pragma unroll
  for (int e = 0; e < NEXP; ++e) { p[e] = __expf(logit[e] - mx); sum += p[e]; }
  float inv = 1.f / sum;

  int i1 = 0; float v1 = -1.f;
#pragma unroll
  for (int e = 0; e < NEXP; ++e) { bool g = p[e] > v1; v1 = g ? p[e] : v1; i1 = g ? e : i1; }
  int i2 = 0; float v2 = -1.f;
#pragma unroll
  for (int e = 0; e < NEXP; ++e) {
    bool g = (e != i1) && (p[e] > v2); v2 = g ? p[e] : v2; i2 = g ? e : i2;
  }
  int e0 = (i1 < i2) ? i1 : i2;
  int e1 = (i1 < i2) ? i2 : i1;
  float w0 = ((e0 == i1) ? v1 : v2) * inv;
  float w1 = ((e1 == i1) ? v1 : v2) * inv;
  tokw[t * 2 + 0] = w0;
  tokw[t * 2 + 1] = w1;

  int s0 = atomicAdd(&cnt[e0], 1);
  btok[e0 * 512 + s0] = t;  bdst[e0 * 512 + s0] = t * 2 + 0;
  int s1 = atomicAdd(&cnt[e1], 1);
  btok[e1 * 512 + s1] = t;  bdst[e1 * 512 + s1] = t * 2 + 1;
}

// ---------------------------------------------------------------------------
// Phase 1 (gate/up projections): h = [time(sp), sp], sp = silu(x@W1^T + b1) *
// (x@W3^T + b3), time = sqrt(||sp||^2 + 1). One workgroup = 16 rows x NCOLS
// cols (NCOLS/64 waves, 4 N-subtiles per wave, K=1024 in steps of 32). LDS row
// layout is pre-shifted: col 0 = time, cols 1.. = sp -> vectorized copy-out.
// BF16W: weights pre-staged bf16 padded to NCOLS rows (no guards in loop).
template <int NCOLS, bool GATHER, bool BF16W>
__global__ __launch_bounds__(NCOLS / 2)
void k_mlp_phase1(const bf16_t* __restrict__ xb,
                  const void* __restrict__ W1v, const float* __restrict__ B1,
                  const void* __restrict__ W3v, const float* __restrict__ B3,
                  bf16_t* __restrict__ hout,
                  const int* __restrict__ cnt, const int* __restrict__ btok) {
  constexpr int NTHR = NCOLS / 2;
  constexpr int NSP  = NCOLS - 1;        // valid output rows of W1/W3
  constexpr int G    = NCOLS / 32;       // reducers per row
  __shared__ __align__(32) float sS[16][NCOLS];
  __shared__ float sRed[16][32];
  __shared__ int   sTok[16];

  const float*  W1f = (const float*)W1v;
  const float*  W3f = (const float*)W3v;
  const bf16_t* W1b = (const bf16_t*)W1v;
  const bf16_t* W3b = (const bf16_t*)W3v;

  int e = 0, mt, n;
  if (GATHER) {
    e = blockIdx.x >> 5;
    mt = blockIdx.x & 31;
    n = cnt[e];
    if (mt * 16 >= n) return;            // uniform per block
    if (BF16W) { W1b += (size_t)e * NCOLS * DIMF; W3b += (size_t)e * NCOLS * DIMF; }
    else       { W1f += (size_t)e * NSP * DIMF;   W3f += (size_t)e * NSP * DIMF;   }
    B1 += (size_t)e * NSP;
    B3 += (size_t)e * NSP;
    hout += (size_t)e * 512 * NCOLS;
  } else {
    mt = blockIdx.x;
    n = T_TOK;
  }
  (void)e;

  int tid = threadIdx.x;
  if (GATHER) {
    if (tid < 16) {
      int slot = mt * 16 + tid;
      sTok[tid] = (slot < n) ? btok[e * 512 + slot] : 0;
    }
    __syncthreads();
  }
  int lane  = tid & 31;
  int wv    = __builtin_amdgcn_readfirstlane(tid >> 5);   // scalarize wave id
  int m     = lane & 15;
  int khalf = (lane >> 4) << 3;          // 0 or 8
  const bf16_t* arow =
      GATHER ? (xb + (size_t)sTok[m] * DIMF) : (xb + (size_t)(mt * 16 + m) * DIMF);

  int n0 = wv * 64;
  const v16bf zf = (v16bf)(bf16_t)(0.0f);
  v8f acc1[4], acc3[4];
#pragma unroll
  for (int c = 0; c < 4; ++c) { acc1[c] = (v8f)(0.f); acc3[c] = (v8f)(0.f); }

  for (int ki = 0; ki < DIMF / 32; ++ki) {
    int k0 = ki * 32 + khalf;
    v16bf va = load_frag_bf16(arow + k0);
#pragma unroll
    for (int c = 0; c < 4; ++c) {
      int nn = n0 + c * 16 + m;
      v16bf vb1, vb3;
      if (BF16W) {                       // padded rows: never out of range
        vb1 = load_frag_bf16(W1b + (size_t)nn * DIMF + k0);
        vb3 = load_frag_bf16(W3b + (size_t)nn * DIMF + k0);
      } else if (n0 + c * 16 + 15 < NSP) {  // scalar fast path: no guard
        vb1 = load_b_f32(W1f + (size_t)nn * DIMF + k0);
        vb3 = load_b_f32(W3f + (size_t)nn * DIMF + k0);
      } else {                           // rare tail: clamp row, zero-select
        int nc = (nn < NSP) ? nn : (NSP - 1);
        vb1 = load_b_f32(W1f + (size_t)nc * DIMF + k0);
        vb3 = load_b_f32(W3f + (size_t)nc * DIMF + k0);
        if (nn >= NSP) { vb1 = zf; vb3 = zf; }
      }
      acc1[c] = WMMA_BF16(va, vb1, acc1[c]);
      acc3[c] = WMMA_BF16(va, vb3, acc3[c]);
    }
  }

  int rbase = (lane >> 4) << 3;          // C layout: rows r + (lane<16?0:8)
#pragma unroll
  for (int c = 0; c < 4; ++c) {
    int nn = n0 + c * 16 + m;
    float b1 = (nn < NSP) ? B1[nn] : 0.f;
    float b3 = (nn < NSP) ? B3[nn] : 0.f;
#pragma unroll
    for (int r = 0; r < 8; ++r) {
      float s1 = acc1[c][r] + b1;
      float a  = s1 / (1.f + __expf(-s1));          // silu
      float sp = a * (acc3[c][r] + b3);
      if (nn < NSP) sS[rbase + r][nn + 1] = sp;     // shifted: col 0 = time
    }
  }
  if (tid < 16) sS[tid][0] = 0.f;
  __syncthreads();

  {
    int r  = tid / G;
    int j0 = tid % G;
    float sum = 0.f;
    for (int j = j0; j < NCOLS; j += G) { float v = sS[r][j]; sum += v * v; }
    sRed[r][j0] = sum;
  }
  __syncthreads();
  if (tid < 16) {
    float tot = 0.f;
#pragma unroll
    for (int j = 0; j < G; ++j) tot += sRed[tid][j];
    sS[tid][0] = sqrtf(fmaxf(tot + 1.0f, 1e-6f));   // _time with clamp
  }
  __syncthreads();

  constexpr int CPR = NCOLS / 8;                    // 8-elem chunks per row
  for (int q = tid; q < 16 * CPR; q += NTHR) {
    int r  = q / CPR;
    int cc = (q % CPR) * 8;
    v8f vf = *(const v8f*)&sS[r][cc];
    v8bf vb = __builtin_convertvector(vf, v8bf);
    int row = mt * 16 + r;
    if (row < n) *(v8bf*)&hout[(size_t)row * NCOLS + cc] = vb;
  }
}

// ---------------------------------------------------------------------------
// Phase 2 (down projection + Lorentz lift): out = [time(s), s],
// s = h@W2^T + b2 (1023 wide). Always N=1024 via 16 waves; K = KDIM.
// BF16W: weights pre-staged bf16 padded to 1024 rows (no guards in loop).
template <int KDIM, bool GATHER, bool BF16W>
__global__ __launch_bounds__(512)
void k_mlp_phase2(const bf16_t* __restrict__ hin,
                  const void* __restrict__ W2v, const float* __restrict__ B2,
                  float* __restrict__ outb,
                  const int* __restrict__ cnt, const int* __restrict__ bdst) {
  __shared__ __align__(32) float sS[16][1024];
  __shared__ float sRed[16][32];

  const float*  W2f = (const float*)W2v;
  const bf16_t* W2b = (const bf16_t*)W2v;

  int e = 0, mt, n;
  if (GATHER) {
    e = blockIdx.x >> 5;
    mt = blockIdx.x & 31;
    n = cnt[e];
    if (mt * 16 >= n) return;
    if (BF16W) W2b += (size_t)e * 1024 * KDIM;
    else       W2f += (size_t)e * SSP * KDIM;
    B2  += (size_t)e * SSP;
    hin += (size_t)e * 512 * KDIM;
  } else {
    mt = blockIdx.x;
    n = T_TOK;
  }
  (void)e;

  int tid = threadIdx.x, lane = tid & 31;
  int wv = __builtin_amdgcn_readfirstlane(tid >> 5);
  int m = lane & 15;
  int khalf = (lane >> 4) << 3;
  const bf16_t* arow = hin + (size_t)(mt * 16 + m) * KDIM;

  int n0 = wv * 64;
  const v16bf zf = (v16bf)(bf16_t)(0.0f);
  v8f acc[4];
#pragma unroll
  for (int c = 0; c < 4; ++c) acc[c] = (v8f)(0.f);

  for (int ki = 0; ki < KDIM / 32; ++ki) {
    int k0 = ki * 32 + khalf;
    v16bf va = load_frag_bf16(arow + k0);
#pragma unroll
    for (int c = 0; c < 4; ++c) {
      int nn = n0 + c * 16 + m;
      v16bf vb;
      if (BF16W) {
        vb = load_frag_bf16(W2b + (size_t)nn * KDIM + k0);
      } else if (n0 + c * 16 + 15 < SSP) {
        vb = load_b_f32(W2f + (size_t)nn * KDIM + k0);
      } else {
        int nc = (nn < SSP) ? nn : (SSP - 1);
        vb = load_b_f32(W2f + (size_t)nc * KDIM + k0);
        if (nn >= SSP) vb = zf;
      }
      acc[c] = WMMA_BF16(va, vb, acc[c]);
    }
  }

  int rbase = (lane >> 4) << 3;
#pragma unroll
  for (int c = 0; c < 4; ++c) {
    int nn = n0 + c * 16 + m;
    float b = (nn < SSP) ? B2[nn] : 0.f;
#pragma unroll
    for (int r = 0; r < 8; ++r)
      if (nn < SSP) sS[rbase + r][nn + 1] = acc[c][r] + b;   // shifted
  }
  if (tid < 16) sS[tid][0] = 0.f;
  __syncthreads();

  {
    int r = tid >> 5, j0 = tid & 31;
    float sum = 0.f;
    for (int j = j0; j < 1024; j += 32) { float v = sS[r][j]; sum += v * v; }
    sRed[r][j0] = sum;
  }
  __syncthreads();
  if (tid < 16) {
    float tot = 0.f;
#pragma unroll
    for (int j = 0; j < 32; ++j) tot += sRed[tid][j];
    sS[tid][0] = sqrtf(fmaxf(tot + 1.0f, 1e-6f));
  }
  __syncthreads();

  for (int q = tid; q < 16 * 256; q += 512) {       // 4-elem chunks per row
    int r  = q >> 8;
    int cc = (q & 255) * 4;
    v4f v = *(const v4f*)&sS[r][cc];
    int row = mt * 16 + r;
    if (row < n) {
      size_t drow = GATHER ? (size_t)bdst[e * 512 + row] : (size_t)row;
      *(v4f*)&outb[drow * 1024 + cc] = v;
    }
  }
}

// ---------------------------------------------------------------------------
// K6: per-token sequential LResNet combine (expert k=0, then k=1, then shared).
// One wave (32 lanes) per token; element idx = i*32 + lane.
__global__ __launch_bounds__(256)
void k_combine(const float* __restrict__ outTok, const float* __restrict__ outSh,
               const float* __restrict__ tokw, float* __restrict__ y) {
  int t    = blockIdx.x * 8 + (threadIdx.x >> 5);
  int lane = threadIdx.x & 31;

  float yv[32];
#pragma unroll
  for (int i = 0; i < 32; ++i) yv[i] = 0.f;
  if (lane == 0) yv[0] = 1.f;                       // Lorentz origin [sqrt(c),0..]

#pragma unroll
  for (int k = 0; k < 2; ++k) {
    const float* o = outTok + ((size_t)t * 2 + k) * 1024;
    float w = tokw[t * 2 + k];
    float ss = 0.f;
#pragma unroll
    for (int i = 0; i < 32; ++i) {
      float z = yv[i] + w * o[i * 32 + lane];
      yv[i] = z;
      ss += z * z;
    }
#pragma unroll
    for (int msk = 16; msk >= 1; msk >>= 1) ss += __shfl_xor(ss, msk, 32);
    float z0 = __shfl(yv[0], 0, 32);
    float rn = 1.f / sqrtf(fmaxf(2.f * z0 * z0 - ss, 1e-6f));   // -inner clamped
#pragma unroll
    for (int i = 0; i < 32; ++i) yv[i] *= rn;
  }
  {
    const float* o = outSh + (size_t)t * 1024;
    float ss = 0.f;
#pragma unroll
    for (int i = 0; i < 32; ++i) {
      float z = yv[i] + o[i * 32 + lane];
      yv[i] = z;
      ss += z * z;
    }
#pragma unroll
    for (int msk = 16; msk >= 1; msk >>= 1) ss += __shfl_xor(ss, msk, 32);
    float z0 = __shfl(yv[0], 0, 32);
    float rn = 1.f / sqrtf(fmaxf(2.f * z0 * z0 - ss, 1e-6f));
#pragma unroll
    for (int i = 0; i < 32; ++i) yv[i] *= rn;
  }
#pragma unroll
  for (int i = 0; i < 32; ++i) y[(size_t)t * 1024 + i * 32 + lane] = yv[i];
}

// ---------------------------------------------------------------------------
extern "C" void kernel_launch(void* const* d_in, const int* in_sizes, int n_in,
                              void* d_out, int out_size, void* d_ws, size_t ws_size,
                              hipStream_t stream) {
  (void)in_sizes; (void)n_in; (void)out_size;
  const float* x   = (const float*)d_in[0];
  const float* gw  = (const float*)d_in[1];
  const float* We1 = (const float*)d_in[2];
  const float* be1 = (const float*)d_in[3];
  const float* We3 = (const float*)d_in[4];
  const float* be3 = (const float*)d_in[5];
  const float* We2 = (const float*)d_in[6];
  const float* be2 = (const float*)d_in[7];
  const float* Ws1 = (const float*)d_in[8];
  const float* bs1 = (const float*)d_in[9];
  const float* Ws3 = (const float*)d_in[10];
  const float* bs3 = (const float*)d_in[11];
  const float* Ws2 = (const float*)d_in[12];
  const float* bs2 = (const float*)d_in[13];

  char* ws = (char*)d_ws;
  bf16_t* xb     = (bf16_t*)(ws + 0);           //  1 MB: x in bf16 [512][1024]
  bf16_t* hR     = (bf16_t*)(ws + 1048576);     //  8 MB: routed h [16][512][512]
  bf16_t* hS     = (bf16_t*)(ws + 9437184);     //  1 MB: shared h [512][1024]
  float*  outTok = (float*) (ws + 10485760);    //  4 MB: per-assignment out [1024][1024]
  float*  outSh  = (float*) (ws + 14680064);    //  2 MB: shared out [512][1024]
  int*    cnt    = (int*)   (ws + 16777216);    //  per-expert counts
  int*    btok   = (int*)   (ws + 16777472);    //  bucket -> token  [16][512]
  int*    bdst   = (int*)   (ws + 16810240);    //  bucket -> t*2+k  [16][512]
  float*  tokw   = (float*) (ws + 16843008);    //  per-token top2 weights [512][2]
  // Staged bf16 weights (zero-padded rows), only used when ws is big enough.
  bf16_t* We1b   = (bf16_t*)(ws + 16847104);    // 16 MB: [16][512][1024]
  bf16_t* We3b   = (bf16_t*)(ws + 33624320);    // 16 MB
  bf16_t* We2b   = (bf16_t*)(ws + 50401536);    // 16 MB: [16][1024][512]
  bf16_t* Ws1b   = (bf16_t*)(ws + 67178752);    //  2 MB: [1024][1024]
  bf16_t* Ws3b   = (bf16_t*)(ws + 69275904);    //  2 MB
  bf16_t* Ws2b   = (bf16_t*)(ws + 71373056);    //  2 MB
  const size_t NEED_FULL = 73470208ull;
  const bool full = (ws_size >= NEED_FULL);     // constant across calls

  k_cvt<<<(T_TOK * DIMF) / (256 * 8), 256, 0, stream>>>(x, xb);
  k_gate<<<1, 512, 0, stream>>>(x, gw, cnt, btok, bdst, tokw);

  if (full) {
    // Stage all weights as zero-padded bf16 (L2-resident for the GEMM pass).
    k_cvt_w< 512, 1024><<<4096, 256, 0, stream>>>(We1, We1b, 511);
    k_cvt_w< 512, 1024><<<4096, 256, 0, stream>>>(We3, We3b, 511);
    k_cvt_w<1024,  512><<<4096, 256, 0, stream>>>(We2, We2b, 1023);
    k_cvt_w<1024, 1024><<< 512, 256, 0, stream>>>(Ws1, Ws1b, 1023);
    k_cvt_w<1024, 1024><<< 512, 256, 0, stream>>>(Ws3, Ws3b, 1023);
    k_cvt_w<1024, 1024><<< 512, 256, 0, stream>>>(Ws2, Ws2b, 1023);

    k_mlp_phase1<RH, true, true><<<NEXP * 32, RH / 2, 0, stream>>>(
        xb, We1b, be1, We3b, be3, hR, cnt, btok);
    k_mlp_phase2<RH, true, true><<<NEXP * 32, 512, 0, stream>>>(
        hR, We2b, be2, outTok, cnt, bdst);
    k_mlp_phase1<SH, false, true><<<32, SH / 2, 0, stream>>>(
        xb, Ws1b, bs1, Ws3b, bs3, hS, nullptr, nullptr);
    k_mlp_phase2<SH, false, true><<<32, 512, 0, stream>>>(
        hS, Ws2b, bs2, outSh, nullptr, nullptr);
  } else {
    // Fallback: convert f32 weights in-register at fragment load.
    k_mlp_phase1<RH, true, false><<<NEXP * 32, RH / 2, 0, stream>>>(
        xb, We1, be1, We3, be3, hR, cnt, btok);
    k_mlp_phase2<RH, true, false><<<NEXP * 32, 512, 0, stream>>>(
        hR, We2, be2, outTok, cnt, bdst);
    k_mlp_phase1<SH, false, false><<<32, SH / 2, 0, stream>>>(
        xb, Ws1, bs1, Ws3, bs3, hS, nullptr, nullptr);
    k_mlp_phase2<SH, false, false><<<32, 512, 0, stream>>>(
        hS, Ws2, bs2, outSh, nullptr, nullptr);
  }

  k_combine<<<T_TOK / 8, 256, 0, stream>>>(outTok, outSh, tokw, (float*)d_out);
}
